// CompetitiveAttention_42949672960449
// MI455X (gfx1250) — matrix operand order repaired and checked by reading, not compiled, not run on hardware
//
#include <hip/hip_runtime.h>

// Problem constants (match reference)
#define BB 8
#define KK 64
#define LL 4096
#define DD 1024
#define HH 16
#define HD 64
#define SCALE 0.125f

// GEMM tiling: 128x128 tile, K-step 32, 256 threads (8 wave32s, 2x4 wave grid)
#define BM 128
#define BN 128
#define BKT 32
#define LDT 40   // padded LDS row in ushort units (80B rows, 16B-aligned)

typedef __attribute__((ext_vector_type(16))) __bf16          v16bf;
typedef __attribute__((ext_vector_type(8)))  float           v8f;
typedef __attribute__((ext_vector_type(8)))  unsigned short  u16x8;
typedef __attribute__((ext_vector_type(4)))  float           f32x4;
typedef __attribute__((ext_vector_type(2)))  unsigned int    u32x2;

// Round-half-up f32->bf16 (epilogue / scalar use).
static __device__ __forceinline__ unsigned short f32_bf16(float f) {
    union { float f; unsigned u; } x; x.f = f;
    return (unsigned short)((x.u + 0x8000u) >> 16);
}

// Pack two f32 into one dword of bf16 (round-half-up): 2x v_add + 1x v_perm.
static __device__ __forceinline__ unsigned pack2_bf16(float f0, float f1) {
    union { float f; unsigned u; } a, b; a.f = f0; b.f = f1;
    unsigned lo = a.u + 0x8000u;
    unsigned hi = b.u + 0x8000u;
    // result bytes {0,1} = lo bytes {2,3}; bytes {2,3} = hi bytes {2,3}
    return __builtin_amdgcn_perm(hi, lo, 0x07060302u);
}

// Build a 16-element bf16 fragment from two 16B LDS chunks (bit-preserving).
static __device__ __forceinline__ v16bf frag16(const unsigned short* p0,
                                               const unsigned short* p1) {
    union { u16x8 u[2]; v16bf v; } c;
    c.u[0] = *(const u16x8*)p0;
    c.u[1] = *(const u16x8*)p1;
    return c.v;
}

// Batched GEMM: C[m,n] = alpha * sum_k A[m,k]*B[n,k] (+ bias[n])
//   A element (m,k): aOff + m*aRow + k                 (f32 if A_F32 else bf16)
//   B element (n,k): bOff + n*bRow + k       (!B_TRANS) (f32 if B_F32 else bf16)
//   B element (n,k): bOff + n + k*bCol       (B_TRANS, bf16 only; n contiguous)
//   C element (m,n): cOff + m*ldc + n                  (f32 if C_F32 else bf16)
// Batch z: (bb = z / Hdim, hh = z % Hdim) with separate strides.
// Requires: KD % 32 == 0; 16B alignment of vector-load offsets; all flat
// element offsets < 2^31 (true for this problem: max ~33.5M elements).
template <bool A_F32, bool B_F32, bool B_TRANS, bool C_F32, bool HAS_BIAS>
__global__ __launch_bounds__(256) void gemm_wmma(
    const void* __restrict__ A, int aRow, int aStrB, int aStrH,
    const void* __restrict__ Bm, int bRow, int bCol, int bStrB, int bStrH,
    void* __restrict__ C, int ldc, int cStrB, int cStrH,
    const float* __restrict__ bias, float alpha,
    int M, int N, int KD, int Hdim)
{
    __shared__ unsigned short As[BM * LDT];
    __shared__ unsigned short Bs[BN * LDT];

    const int tid  = threadIdx.x;
    const int lane = tid & 31;
    const int wave = tid >> 5;
    const int wm   = wave >> 2;       // 0..1
    const int wn   = wave & 3;        // 0..3
    const int half = lane >> 4;       // 0..1
    const int lr   = lane & 15;       // 0..15

    const int bh = blockIdx.z;
    const int bb = bh / Hdim;
    const int hh = bh - bb * Hdim;
    const int aOff = bb * aStrB + hh * aStrH;
    const int bOff = bb * bStrB + hh * bStrH;
    const int cOff = bb * cStrB + hh * cStrH;

    const int tileM = (int)blockIdx.y * BM;
    const int tileN = (int)blockIdx.x * BN;

    const float*          Af = (const float*)A;
    const unsigned short* Ab = (const unsigned short*)A;
    const float*          Bf = (const float*)Bm;
    const unsigned short* Bb = (const unsigned short*)Bm;

    v8f acc[4][2];
    #pragma unroll
    for (int i = 0; i < 4; ++i)
        #pragma unroll
        for (int j = 0; j < 2; ++j)
            acc[i][j] = (v8f){0.f, 0.f, 0.f, 0.f, 0.f, 0.f, 0.f, 0.f};

    // Register staging buffers (unused ones eliminated per instantiation).
    f32x4 aRegF[4], bRegF[4];
    u16x8 aRegB[2], bRegB[2];

    // ---- global -> register tile load for K-offset k0 ----
    auto load_regs = [&](int k0) {
        if (A_F32) {
            #pragma unroll
            for (int q = 0; q < 4; ++q) {
                int ch = q * 256 + tid;           // 1024 float4 chunks
                int r  = ch >> 3;
                int c  = (ch & 7) << 2;
                int gm = tileM + r;
                f32x4 v = {0.f, 0.f, 0.f, 0.f};
                if (gm < M) v = *(const f32x4*)(Af + aOff + gm * aRow + k0 + c);
                aRegF[q] = v;
            }
        } else {
            #pragma unroll
            for (int q = 0; q < 2; ++q) {
                int ch = q * 256 + tid;           // 512 u16x8 chunks
                int r  = ch >> 2;
                int c  = (ch & 3) << 3;
                int gm = tileM + r;
                u16x8 v = {0, 0, 0, 0, 0, 0, 0, 0};
                if (gm < M) v = *(const u16x8*)(Ab + aOff + gm * aRow + k0 + c);
                aRegB[q] = v;
            }
        }
        if (B_TRANS) {
            // bf16, contiguous along n: addr = bOff + n + k*bCol
            #pragma unroll
            for (int q = 0; q < 2; ++q) {
                int ch   = q * 256 + tid;         // 512 chunks: 32 k x 16 n-octets
                int kcol = ch >> 4;
                int n8   = (ch & 15) << 3;
                int gn   = tileN + n8;
                u16x8 v = {0, 0, 0, 0, 0, 0, 0, 0};
                if (gn < N)
                    v = *(const u16x8*)(Bb + bOff + (k0 + kcol) * bCol + gn);
                bRegB[q] = v;
            }
        } else if (B_F32) {
            #pragma unroll
            for (int q = 0; q < 4; ++q) {
                int ch = q * 256 + tid;
                int r  = ch >> 3;
                int c  = (ch & 7) << 2;
                int gn = tileN + r;
                f32x4 v = {0.f, 0.f, 0.f, 0.f};
                if (gn < N) v = *(const f32x4*)(Bf + bOff + gn * bRow + k0 + c);
                bRegF[q] = v;
            }
        } else {
            #pragma unroll
            for (int q = 0; q < 2; ++q) {
                int ch = q * 256 + tid;
                int r  = ch >> 2;
                int c  = (ch & 3) << 3;
                int gn = tileN + r;
                u16x8 v = {0, 0, 0, 0, 0, 0, 0, 0};
                if (gn < N) v = *(const u16x8*)(Bb + bOff + gn * bRow + k0 + c);
                bRegB[q] = v;
            }
        }
    };

    // ---- register -> LDS tile store (f32 converted via packed perm) ----
    auto store_lds = [&]() {
        if (A_F32) {
            #pragma unroll
            for (int q = 0; q < 4; ++q) {
                int ch = q * 256 + tid;
                int r  = ch >> 3;
                int c  = (ch & 7) << 2;
                u32x2 s = { pack2_bf16(aRegF[q][0], aRegF[q][1]),
                            pack2_bf16(aRegF[q][2], aRegF[q][3]) };
                *(u32x2*)&As[r * LDT + c] = s;
            }
        } else {
            #pragma unroll
            for (int q = 0; q < 2; ++q) {
                int ch = q * 256 + tid;
                int r  = ch >> 2;
                int c  = (ch & 3) << 3;
                *(u16x8*)&As[r * LDT + c] = aRegB[q];
            }
        }
        if (B_TRANS) {
            #pragma unroll
            for (int q = 0; q < 2; ++q) {
                int ch   = q * 256 + tid;
                int kcol = ch >> 4;
                int n8   = (ch & 15) << 3;
                #pragma unroll
                for (int t = 0; t < 8; ++t)
                    Bs[(n8 + t) * LDT + kcol] = bRegB[q][t];
            }
        } else if (B_F32) {
            #pragma unroll
            for (int q = 0; q < 4; ++q) {
                int ch = q * 256 + tid;
                int r  = ch >> 3;
                int c  = (ch & 7) << 2;
                u32x2 s = { pack2_bf16(bRegF[q][0], bRegF[q][1]),
                            pack2_bf16(bRegF[q][2], bRegF[q][3]) };
                *(u32x2*)&Bs[r * LDT + c] = s;
            }
        } else {
            #pragma unroll
            for (int q = 0; q < 2; ++q) {
                int ch = q * 256 + tid;
                int r  = ch >> 2;
                int c  = (ch & 3) << 3;
                *(u16x8*)&Bs[r * LDT + c] = bRegB[q];
            }
        }
    };

    load_regs(0);

    for (int k0 = 0; k0 < KD; k0 += BKT) {
        __syncthreads();            // previous compute done before LDS overwrite
        store_lds();
        __syncthreads();

        // Issue next tile's global loads early; WMMA block below hides latency.
        if (k0 + BKT < KD) load_regs(k0 + BKT);

        // Speculative L2 prefetch two tiles ahead on the A stream.
        if (k0 + 2 * BKT < KD) {
            int gm = tileM + (tid >> 1);
            if (gm < M) {
                int off = aOff + gm * aRow + k0 + 2 * BKT + (tid & 1) * 16;
                __builtin_prefetch(A_F32 ? (const void*)(Af + off)
                                         : (const void*)(Ab + off), 0, 1);
            }
        }

        // ---- fragments per documented 16-bit A/B VGPR layouts + 8x WMMA ----
        v16bf afr[4];
        #pragma unroll
        for (int i = 0; i < 4; ++i) {
            int row = wm * 64 + i * 16 + lr;
            const unsigned short* p = &As[row * LDT + half * 8];  // K 0..7 / 8..15
            afr[i] = frag16(p, p + 16);                           // K 16..23 / 24..31
        }
        v16bf bfr[2];
        #pragma unroll
        for (int j = 0; j < 2; ++j) {
            int nrow = wn * 32 + j * 16 + lr;
            const unsigned short* p = &Bs[nrow * LDT + half * 8];  // K 0..7 / 8..15
            bfr[j] = frag16(p, p + 16);                            // K 16..23 / 24..31
        }
        #pragma unroll
        for (int i = 0; i < 4; ++i)
            #pragma unroll
            for (int j = 0; j < 2; ++j)
                acc[i][j] = __builtin_amdgcn_wmma_f32_16x16x32_bf16(
                    false, afr[i], false, bfr[j], (short)0, acc[i][j],
                    false, false);
    }

    // ---- epilogue: alpha, bias, store per documented C layout ----
    float*          Cf = (float*)C;
    unsigned short* Cb = (unsigned short*)C;
    #pragma unroll
    for (int i = 0; i < 4; ++i) {
        int mBase = tileM + wm * 64 + i * 16 + half * 8;
        #pragma unroll
        for (int j = 0; j < 2; ++j) {
            int n = tileN + wn * 32 + j * 16 + lr;
            if (n >= N) continue;
            float bsv = HAS_BIAS ? bias[n] : 0.f;
            #pragma unroll
            for (int r = 0; r < 8; ++r) {
                int m = mBase + r;
                if (m < M) {
                    float v = alpha * acc[i][j][r] + bsv;
                    int off = cOff + m * ldc + n;
                    if (C_F32) Cf[off] = v;
                    else       Cb[off] = f32_bf16(v);
                }
            }
        }
    }
}

// Softmax over the slot axis K (axis=-2 of [B,H,K,L]): each thread owns one
// (b,h,l) column of 64 strided floats; loads are coalesced across l.
__global__ __launch_bounds__(256) void softmax_slots(float* __restrict__ attn)
{
    int idx = (int)blockIdx.x * 256 + (int)threadIdx.x;
    const int total = BB * HH * LL;
    if (idx >= total) return;
    int bhv = idx >> 12;          // / LL
    int l   = idx & (LL - 1);
    float* base = attn + (long long)bhv * KK * LL + l;

    float v[KK];
    float mx = -3.4e38f;
    #pragma unroll
    for (int k = 0; k < KK; ++k) {
        v[k] = base[k * LL];
        mx = fmaxf(mx, v[k]);
    }
    float s = 0.f;
    #pragma unroll
    for (int k = 0; k < KK; ++k) {
        v[k] = __expf(v[k] - mx);
        s += v[k];
    }
    float inv = 1.f / s;
    #pragma unroll
    for (int k = 0; k < KK; ++k)
        base[k * LL] = v[k] * inv;
}

// Renormalize each (b,h,k) row over L: attn /= (sum_l attn + 1e-8)
__global__ __launch_bounds__(256) void renorm_rows(float* __restrict__ attn)
{
    float* p = attn + (long long)blockIdx.x * LL;   // block = one (b,h,k) row
    float s = 0.f;
    for (int i = threadIdx.x; i < LL; i += 256) s += p[i];
    __shared__ float red[256];
    red[threadIdx.x] = s;
    __syncthreads();
    #pragma unroll
    for (int st = 128; st > 0; st >>= 1) {
        if (threadIdx.x < st) red[threadIdx.x] += red[threadIdx.x + st];
        __syncthreads();
    }
    float inv = 1.f / (red[0] + 1e-8f);
    for (int i = threadIdx.x; i < LL; i += 256) p[i] *= inv;
}

extern "C" void kernel_launch(void* const* d_in, const int* in_sizes, int n_in,
                              void* d_out, int out_size, void* d_ws, size_t ws_size,
                              hipStream_t stream)
{
    const float* query = (const float*)d_in[0];
    const float* key   = (const float*)d_in[1];
    const float* value = (const float*)d_in[2];
    const float* Wq    = (const float*)d_in[3];
    const float* Wk    = (const float*)d_in[4];
    const float* Wv    = (const float*)d_in[5];
    const float* Wo    = (const float*)d_in[6];
    const float* bo    = (const float*)d_in[7];

    float* out  = (float*)d_out;                       // [B,K,D] f32
    float* attn = out + (long long)BB * KK * DD;       // [B,H,K,L] f32

    // bf16 scratch (ushort bit-storage): qp, kp, vp, ctx
    unsigned short* qp  = (unsigned short*)d_ws;
    unsigned short* kp  = qp + (long long)BB * KK * DD;
    unsigned short* vp  = kp + (long long)BB * LL * DD;
    unsigned short* ctx = vp + (long long)BB * LL * DD;

    dim3 blk(256);

    // 1) qp = query @ Wq^T  (M=512, N=1024, K=1024) -> bf16
    gemm_wmma<true, true, false, false, false>
        <<<dim3(DD / BN, (BB * KK) / BM, 1), blk, 0, stream>>>(
        query, DD, 0, 0,
        Wq,    DD, 1, 0, 0,
        qp,    DD, 0, 0,
        nullptr, 1.0f, BB * KK, DD, DD, 1);

    // 2) kp = key @ Wk^T    (M=32768, N=1024, K=1024) -> bf16
    gemm_wmma<true, true, false, false, false>
        <<<dim3(DD / BN, (BB * LL) / BM, 1), blk, 0, stream>>>(
        key, DD, 0, 0,
        Wk,  DD, 1, 0, 0,
        kp,  DD, 0, 0,
        nullptr, 1.0f, BB * LL, DD, DD, 1);

    // 3) vp = value @ Wv^T  (M=32768, N=1024, K=1024) -> bf16
    gemm_wmma<true, true, false, false, false>
        <<<dim3(DD / BN, (BB * LL) / BM, 1), blk, 0, stream>>>(
        value, DD, 0, 0,
        Wv,    DD, 1, 0, 0,
        vp,    DD, 0, 0,
        nullptr, 1.0f, BB * LL, DD, DD, 1);

    // 4) dots[b,h] = SCALE * qp[b,:,h] @ kp[b,:,h]^T  (M=64, N=4096, K=64) x 128
    //    straight into the attn region of d_out (f32)
    gemm_wmma<false, false, false, true, false>
        <<<dim3(LL / BN, 1, BB * HH), blk, 0, stream>>>(
        qp, DD, KK * DD, HD,
        kp, DD, 1, LL * DD, HD,
        attn, LL, HH * KK * LL, KK * LL,
        nullptr, SCALE, KK, LL, HD, HH);

    // 5) softmax over slot axis K, in place
    softmax_slots<<<dim3((BB * HH * LL + 255) / 256), blk, 0, stream>>>(attn);

    // 6) renormalize over keys L, in place
    renorm_rows<<<dim3(BB * HH * KK), blk, 0, stream>>>(attn);

    // 7) ctx[b,h] = attn[b,h] @ vp[b,:,h]  (M=64, N=64, K=4096) x 128 -> bf16
    //    A is f32 (attn); B transposed access (n contiguous, k stride = D)
    gemm_wmma<true, false, true, false, false>
        <<<dim3(1, 1, BB * HH), blk, 0, stream>>>(
        attn, LL, HH * KK * LL, KK * LL,
        vp,   1, DD, LL * DD, HD,
        ctx,  DD, KK * DD, HD,
        nullptr, 1.0f, KK, HD, LL, HH);

    // 8) out = ctx @ Wo^T + bo  (M=512, N=1024, K=1024) -> f32 into d_out
    gemm_wmma<false, true, false, true, true>
        <<<dim3(DD / BN, (BB * KK) / BM, 1), blk, 0, stream>>>(
        ctx, DD, 0, 0,
        Wo,  DD, 1, 0, 0,
        out, DD, 0, 0,
        bo, 1.0f, BB * KK, DD, DD, 1);
}
